// Model15_9620726743230
// MI455X (gfx1250) — matrix operand order repaired and checked by reading, not compiled, not run on hardware
//
#include <hip/hip_runtime.h>
#include <hip/hip_bf16.h>
#include <math.h>

typedef __attribute__((ext_vector_type(16))) _Float16 v16h;
typedef __attribute__((ext_vector_type(8)))  float    v8f;

#define N_NODES  131072
#define N_EDGES  4194304
#define NB_NODES 65536
#define EB_EDGES 1048576
#define NBON_    8192
#define NNZ_     131072
#define BATCH_   256

// ---------------- float <-> monotonic u32 mapping for atomic max ----------------
__device__ __forceinline__ unsigned fmapU(float f) {
  unsigned u = __float_as_uint(f);
  return (u & 0x80000000u) ? ~u : (u | 0x80000000u);
}
__device__ __forceinline__ float funmapU(unsigned m) {
  unsigned u = (m & 0x80000000u) ? (m & 0x7FFFFFFFu) : ~m;
  return __uint_as_float(u);
}

// ---------------- fills ----------------
__global__ void fill_u32_k(unsigned* p, int n, unsigned v) {
  int i = blockIdx.x * blockDim.x + threadIdx.x;
  if (i < n) p[i] = v;
}
__global__ void fill_f32_k(float* p, int n, float v) {
  int i = blockIdx.x * blockDim.x + threadIdx.x;
  if (i < n) p[i] = v;
}

// ---------------- WMMA projection: up to 4 simultaneous linear layers ----------------
// O_w = X @ W_w + b_w (optionally ReLU) via v_wmma_f32_16x16x32_f16.
// DIN/DOUT are compile-time so all fragment loops fully unroll (no indexed VGPRs),
// and weights are staged in LDS in *fragment-ready* order: the 16 halfs each lane
// needs for a B fragment are contiguous -> vector ds_load instead of 16x ds_load_u16.
template<int NW, bool RELU, int DIN, int DOUT>
__global__ void __launch_bounds__(256) proj_wmma(
    const float* __restrict__ X, int ldx, int nrows,
    const float* __restrict__ W0, const float* __restrict__ B0,
    const float* __restrict__ W1, const float* __restrict__ B1,
    const float* __restrict__ W2, const float* __restrict__ B2,
    const float* __restrict__ W3, const float* __restrict__ B3,
    float* __restrict__ O0, float* __restrict__ O1,
    float* __restrict__ O2, float* __restrict__ O3, int ldo)
{
  constexpr int NK = (DIN + 31) / 32;    // K-steps of 32
  constexpr int NC = (DOUT + 15) / 16;   // N tiles of 16
  constexpr int NFRAG = NW * NC * NK;

  // sWf[frag][lane][16 halfs]  -- 32B per lane, 32B aligned
  __shared__ __align__(32) _Float16 sWf[NFRAG * 32 * 16];
  __shared__ float sB[NW][64];
  const float* Ws[4] = {W0, W1, W2, W3};
  const float* Bs[4] = {B0, B1, B2, B3};
  float*       Os[4] = {O0, O1, O2, O3};

  // cooperative fragment-ready staging (f32 -> f16)
  for (int idx = threadIdx.x; idx < NFRAG * 512; idx += blockDim.x) {
    int t = idx & 15;             // half within lane's fragment
    int l = (idx >> 4) & 31;      // lane
    int f = idx >> 9;             // fragment index
    int kk = f % NK;
    int c  = (f / NK) % NC;
    int w  = f / (NK * NC);
    int col = c * 16 + (l & 15);
    int k   = kk * 32 + ((l >> 4) << 3) + (t & 7) + ((t >> 3) << 4);   // B layout: lanes 0-15 K 0..7,16..23; 16-31 K 8..15,24..31
    float v = (k < DIN && col < DOUT) ? Ws[w][k * DOUT + col] : 0.0f;
    sWf[idx] = (_Float16)v;
  }
  for (int idx = threadIdx.x; idx < NW * 64; idx += blockDim.x) {
    int w = idx >> 6, c = idx & 63;
    sB[w][c] = (c < DOUT) ? Bs[w][c] : 0.0f;
  }
  __syncthreads();

  const int lane = threadIdx.x & 31;
  const int wave = blockIdx.x * (blockDim.x >> 5) + (threadIdx.x >> 5);
  const int r0 = wave * 16;
  if (r0 >= nrows) return;

  // ---- A fragments (ISA 7.12.2 16-bit A 16x32 layout), fully unrolled ----
  const int hi = (lane >> 4) << 3;            // 0 (lanes 0-15) or 8 (lanes 16-31)
  const int arow = r0 + (lane & 15);
  const float* xr = X + (size_t)arow * ldx;
  v16h afrag[NK];
#pragma unroll
  for (int kk = 0; kk < NK; ++kk) {
    v16h a;
#pragma unroll
    for (int j = 0; j < 4; ++j) {            // VGPRs 0..3 : K = base+2j, base+2j+1
      int k0 = kk * 32 + hi + 2 * j;
      a[2 * j]     = (_Float16)((k0     < DIN) ? xr[k0]     : 0.0f);
      a[2 * j + 1] = (_Float16)((k0 + 1 < DIN) ? xr[k0 + 1] : 0.0f);
    }
#pragma unroll
    for (int j = 0; j < 4; ++j) {            // VGPRs 4..7 : K = base+16+2j, +1
      int k0 = kk * 32 + hi + 16 + 2 * j;
      a[8 + 2 * j]     = (_Float16)((k0     < DIN) ? xr[k0]     : 0.0f);
      a[8 + 2 * j + 1] = (_Float16)((k0 + 1 < DIN) ? xr[k0 + 1] : 0.0f);
    }
    afrag[kk] = a;
  }

  const int ccol = lane & 15;
  const int crow = r0 + ((lane >> 4) << 3);   // C/D: rows r0..r0+7 or r0+8..r0+15

#pragma unroll
  for (int w = 0; w < NW; ++w) {
#pragma unroll
    for (int c = 0; c < NC; ++c) {
      v8f acc = {};
#pragma unroll
      for (int kk = 0; kk < NK; ++kk) {
        const v16h bfrag =
            *(const v16h*)&sWf[(((w * NC + c) * NK + kk) * 32 + lane) * 16];
        acc = __builtin_amdgcn_wmma_f32_16x16x32_f16(
            false, afrag[kk], false, bfrag, (short)0, acc, false, false);
      }
      const int col = c * 16 + ccol;
      if (col < DOUT) {
        const float bias = sB[w][col];
        float* op = Os[w] + (size_t)crow * ldo + col;
#pragma unroll
        for (int i = 0; i < 8; ++i) {
          float v = acc[i] + bias;
          if (RELU) v = v > 0.0f ? v : 0.0f;
          op[(size_t)i * ldo] = v;
        }
      }
    }
  }
}

// ---------------- gather rows ----------------
__global__ void gather_rows(const float* __restrict__ X, int ldx,
                            const int* __restrict__ idx,
                            float* __restrict__ out, int ldo, int n, int d) {
  int i = blockIdx.x * blockDim.x + threadIdx.x;
  if (i >= n) return;
  const float* s = X + (size_t)idx[i] * ldx;
  float* o = out + (size_t)i * ldo;
  for (int j = 0; j < d; ++j) o[j] = s[j];
}

// ---------------- edge attention passes ----------------
__global__ void edge_score_max(const int* __restrict__ edges, int ne,
                               const float* __restrict__ Q, const float* __restrict__ K,
                               int ld, int d, float inv_sqrt_d,
                               float* __restrict__ score, unsigned* __restrict__ smax) {
  int e = blockIdx.x * blockDim.x + threadIdx.x;
  if (e >= ne) return;
  int src = edges[e];
  int dst = edges[ne + e];
  const float* q = Q + (size_t)dst * ld;
  const float* k = K + (size_t)src * ld;
  float s = 0.0f;
  for (int j = 0; j < d; ++j) s += q[j] * k[j];
  s *= inv_sqrt_d;
  score[e] = s;
  atomicMax(&smax[dst], fmapU(s));
}

__global__ void edge_exp_sum(const int* __restrict__ edges, int ne,
                             float* __restrict__ score,
                             const unsigned* __restrict__ smax,
                             float* __restrict__ denom) {
  int e = blockIdx.x * blockDim.x + threadIdx.x;
  if (e >= ne) return;
  int dst = edges[ne + e];
  float w = expf(score[e] - funmapU(smax[dst]));
  score[e] = w;
  atomicAdd(&denom[dst], w);
}

__global__ void edge_scatter(const int* __restrict__ edges, int ne,
                             const float* __restrict__ score,
                             const float* __restrict__ denom,
                             const float* __restrict__ V, int ldv, int d,
                             float* __restrict__ O, int ldo) {
  int e = blockIdx.x * blockDim.x + threadIdx.x;
  if (e >= ne) return;
  int src = edges[e];
  int dst = edges[ne + e];
  float a = score[e] / denom[dst];
  const float* v = V + (size_t)src * ldv;
  float* o = O + (size_t)dst * ldo;
  for (int j = 0; j < d; ++j) atomicAdd(&o[j], a * v[j]);
}

// ---------------- beta-gate combine + ReLU ----------------
__global__ void combine_relu(const float* __restrict__ O, int ldo_,
                             const float* __restrict__ S, int lds_,
                             int n, int d, const float* __restrict__ Wb,
                             float* __restrict__ out, int ldout) {
  int i = blockIdx.x * blockDim.x + threadIdx.x;
  if (i >= n) return;
  const float* o = O + (size_t)i * ldo_;
  const float* s = S + (size_t)i * lds_;
  float g = 0.0f;
  for (int j = 0; j < d; ++j) {
    float ov = o[j], sv = s[j];
    g += ov * Wb[j] + sv * Wb[d + j] + (ov - sv) * Wb[2 * d + j];
  }
  float beta = 1.0f / (1.0f + expf(-g));
  float* p = out + (size_t)i * ldout;
  for (int j = 0; j < d; ++j) {
    float r = beta * s[j] + (1.0f - beta) * o[j];
    p[j] = r > 0.0f ? r : 0.0f;
  }
}

// ---------------- bonus pooling / spmm / concat / batch pool ----------------
__global__ void pool_bonus(const float* __restrict__ Bv, int ldb,
                           const int* __restrict__ seg,
                           float* __restrict__ pbv, int ldp, int n, int d) {
  int i = blockIdx.x * blockDim.x + threadIdx.x;
  if (i >= n) return;
  int b = seg[i];
  const float* s = Bv + (size_t)i * ldb;
  float* p = pbv + (size_t)b * ldp;
  for (int j = 0; j < d; ++j) atomicAdd(&p[j], s[j]);
}

__global__ void spmm_kernel(const int* __restrict__ map, int nnz,
                            const float* __restrict__ vals,
                            const float* __restrict__ mat, int ldm,
                            float* __restrict__ out, int ldo, int d) {
  int i = blockIdx.x * blockDim.x + threadIdx.x;
  if (i >= nnz) return;
  int row = map[i];
  int col = map[nnz + i];
  float v = vals[col];
  const float* m = mat + (size_t)col * ldm;
  float* o = out + (size_t)row * ldo;
  for (int j = 0; j < d; ++j) atomicAdd(&o[j], v * m[j]);
}

__global__ void concat_kernel(const float* __restrict__ A, int lda,
                              const float* __restrict__ Bm, int ldb,
                              float* __restrict__ Cc, int ldc,
                              int n, int da, int db) {
  int i = blockIdx.x * blockDim.x + threadIdx.x;
  if (i >= n) return;
  const float* a = A + (size_t)i * lda;
  const float* b = Bm + (size_t)i * ldb;
  float* c = Cc + (size_t)i * ldc;
  for (int j = 0; j < da; ++j) c[j] = a[j];
  for (int j = 0; j < db; ++j) c[da + j] = b[j];
}

__global__ void pool_graph(const float* __restrict__ X, int ldx,
                           const int* __restrict__ bb,
                           float* __restrict__ psum, float* __restrict__ pcnt,
                           int n, int d) {
  int i = blockIdx.x * blockDim.x + threadIdx.x;
  if (i >= n) return;
  int b = bb[i];
  const float* x = X + (size_t)i * ldx;
  float* p = psum + (size_t)b * 64;
  for (int j = 0; j < d; ++j) atomicAdd(&p[j], x[j]);
  atomicAdd(&pcnt[b], 1.0f);
}

// ---------------- final MLP head ----------------
__global__ void head_kernel(const float* __restrict__ psum, const float* __restrict__ pcnt,
                            const float* __restrict__ income, const float* __restrict__ armies,
                            const float* __restrict__ W1, const float* __restrict__ b1,
                            const float* __restrict__ W2, const float* __restrict__ b2,
                            float* __restrict__ out) {
  int b = threadIdx.x;
  if (b >= BATCH_) return;
  float in[65];
  float c = pcnt[b];
  if (c < 1.0f) c = 1.0f;
  for (int j = 0; j < 60; ++j) in[j] = psum[b * 64 + j] / c;
  for (int j = 0; j < 4; ++j) in[60 + j] = income[b * 4 + j];
  in[64] = armies[b];
  float h[60];
  for (int k = 0; k < 60; ++k) {
    float a = b1[k];
    for (int j = 0; j < 65; ++j) a += in[j] * W1[j * 60 + k];
    h[k] = a > 0.0f ? a : 0.0f;
  }
  float v = b2[0];
  for (int k = 0; k < 60; ++k) v += h[k] * W2[k];
  out[b] = tanhf(v);
}

// =====================================================================
extern "C" void kernel_launch(void* const* d_in, const int* in_sizes, int n_in,
                              void* d_out, int out_size, void* d_ws, size_t ws_size,
                              hipStream_t stream) {
  (void)in_sizes; (void)n_in; (void)ws_size;
  const float* graph_features = (const float*)d_in[0];
  const int*   graph_edges    = (const int*)d_in[1];
  const int*   bonus_nodes    = (const int*)d_in[2];
  const int*   bonus_edges    = (const int*)d_in[3];
  const int*   bonus_batch    = (const int*)d_in[4];
  const int*   bonus_mapping  = (const int*)d_in[5];
  const float* bonus_values   = (const float*)d_in[6];
  const int*   batch          = (const int*)d_in[7];
  const float* income         = (const float*)d_in[8];
  const float* armies         = (const float*)d_in[9];
  // d_in[10] = num_moves scalar (value recovered host-side from out_size)
  auto P = [&](int i) { return (const float*)d_in[i]; };
  const float* Wi  = P(11); const float* bi  = P(12);
  const float* W1f = P(67); const float* b1f = P(68);
  const float* W2f = P(69); const float* b2f = P(70);

  // ---- carve workspace ----
  char* w = (char*)d_ws;
  size_t off = 0;
  auto carve = [&](size_t bytes) -> void* {
    void* p = w + off;
    off += (bytes + 255) & ~(size_t)255;
    return p;
  };
  float*    X     = (float*)carve((size_t)N_NODES * 64 * 4);
  float*    XC    = (float*)carve((size_t)N_NODES * 64 * 4);
  float*    Q     = (float*)carve((size_t)N_NODES * 64 * 4);
  float*    Kb    = (float*)carve((size_t)N_NODES * 64 * 4);
  float*    V     = (float*)carve((size_t)N_NODES * 64 * 4);
  float*    S     = (float*)carve((size_t)N_NODES * 64 * 4);
  float*    O     = (float*)carve((size_t)N_NODES * 64 * 4);
  float*    Bin   = (float*)carve((size_t)NB_NODES * 32 * 4);
  float*    score = (float*)carve((size_t)N_EDGES * 4);
  unsigned* smax  = (unsigned*)carve((size_t)N_NODES * 4);
  float*    denom = (float*)carve((size_t)N_NODES * 4);
  float*    pb    = (float*)carve((size_t)NBON_ * 32 * 4);
  float*    bsc   = (float*)carve((size_t)N_NODES * 32 * 4);
  float*    psum  = (float*)carve((size_t)BATCH_ * 64 * 4);
  float*    pcnt  = (float*)carve((size_t)BATCH_ * 4);

  const unsigned NEG_INF_MAP = 0x007FFFFFu;  // fmapU(-inf)
  dim3 blk(256);

  // ---- init projection: X = relu(GF @ Wi + bi), N x 5 -> N x 20 ----
  proj_wmma<1, true, 5, 20><<<N_NODES / 128, 256, 0, stream>>>(
      graph_features, 5, N_NODES,
      Wi, bi, nullptr, nullptr, nullptr, nullptr, nullptr, nullptr,
      X, nullptr, nullptr, nullptr, 64);

  const int base_b[3] = {13, 31, 49};  // b1, b2, b3 (q.W,q.b,k.W,k.b,v.W,v.b,skip.W,skip.b,Wb)
  const int base_g[3] = {22, 40, 58};  // g1, g2, g3
  for (int r = 0; r < 3; ++r) {
    const int doutg = (r == 2) ? 60 : 20;

    // ---- bonus branch: gather -> tconv (NB nodes, EB edges, 20 -> 20) ----
    gather_rows<<<NB_NODES / 256, blk, 0, stream>>>(X, 64, bonus_nodes, Bin, 32, NB_NODES, 20);
    {
      int t = base_b[r];
      proj_wmma<4, false, 20, 20><<<NB_NODES / 128, 256, 0, stream>>>(
          Bin, 32, NB_NODES,
          P(t + 0), P(t + 1), P(t + 2), P(t + 3),
          P(t + 4), P(t + 5), P(t + 6), P(t + 7),
          Q, Kb, V, S, 64);
      hipMemsetAsync(O, 0, (size_t)NB_NODES * 64 * 4, stream);
      fill_u32_k<<<NB_NODES / 256, blk, 0, stream>>>(smax, NB_NODES, NEG_INF_MAP);
      hipMemsetAsync(denom, 0, (size_t)NB_NODES * 4, stream);
      edge_score_max<<<EB_EDGES / 256, blk, 0, stream>>>(
          bonus_edges, EB_EDGES, Q, Kb, 64, 20, 1.0f / sqrtf(20.0f), score, smax);
      edge_exp_sum<<<EB_EDGES / 256, blk, 0, stream>>>(bonus_edges, EB_EDGES, score, smax, denom);
      edge_scatter<<<EB_EDGES / 256, blk, 0, stream>>>(
          bonus_edges, EB_EDGES, score, denom, V, 64, 20, O, 64);
      combine_relu<<<NB_NODES / 256, blk, 0, stream>>>(O, 64, S, 64, NB_NODES, 20, P(t + 8), Bin, 32);
    }
    // ---- global_add_pool over bonuses, then spmm scatter back to nodes ----
    hipMemsetAsync(pb, 0, (size_t)NBON_ * 32 * 4, stream);
    pool_bonus<<<NB_NODES / 256, blk, 0, stream>>>(Bin, 32, bonus_batch, pb, 32, NB_NODES, 20);
    hipMemsetAsync(bsc, 0, (size_t)N_NODES * 32 * 4, stream);
    spmm_kernel<<<NNZ_ / 256, blk, 0, stream>>>(bonus_mapping, NNZ_, bonus_values, pb, 32, bsc, 32, 20);
    concat_kernel<<<N_NODES / 256, blk, 0, stream>>>(X, 64, bsc, 32, XC, 64, N_NODES, 20, 20);

    // ---- graph tconv: N nodes, E edges, 40 -> doutg ----
    {
      int t = base_g[r];
      if (doutg == 20) {
        proj_wmma<4, false, 40, 20><<<N_NODES / 128, 256, 0, stream>>>(
            XC, 64, N_NODES,
            P(t + 0), P(t + 1), P(t + 2), P(t + 3),
            P(t + 4), P(t + 5), P(t + 6), P(t + 7),
            Q, Kb, V, S, 64);
      } else {
        proj_wmma<4, false, 40, 60><<<N_NODES / 128, 256, 0, stream>>>(
            XC, 64, N_NODES,
            P(t + 0), P(t + 1), P(t + 2), P(t + 3),
            P(t + 4), P(t + 5), P(t + 6), P(t + 7),
            Q, Kb, V, S, 64);
      }
      hipMemsetAsync(O, 0, (size_t)N_NODES * 64 * 4, stream);
      fill_u32_k<<<N_NODES / 256, blk, 0, stream>>>(smax, N_NODES, NEG_INF_MAP);
      hipMemsetAsync(denom, 0, (size_t)N_NODES * 4, stream);
      edge_score_max<<<N_EDGES / 256, blk, 0, stream>>>(
          graph_edges, N_EDGES, Q, Kb, 64, doutg, 1.0f / sqrtf((float)doutg), score, smax);
      edge_exp_sum<<<N_EDGES / 256, blk, 0, stream>>>(graph_edges, N_EDGES, score, smax, denom);
      edge_scatter<<<N_EDGES / 256, blk, 0, stream>>>(
          graph_edges, N_EDGES, score, denom, V, 64, doutg, O, 64);
      combine_relu<<<N_NODES / 256, blk, 0, stream>>>(O, 64, S, 64, N_NODES, doutg, P(t + 8), X, 64);
    }
  }

  // ---- readout: batch mean pool + MLP head ----
  hipMemsetAsync(psum, 0, (size_t)BATCH_ * 64 * 4, stream);
  hipMemsetAsync(pcnt, 0, (size_t)BATCH_ * 4, stream);
  pool_graph<<<N_NODES / 256, blk, 0, stream>>>(X, 64, batch, psum, pcnt, N_NODES, 60);
  float* out = (float*)d_out;
  head_kernel<<<1, 256, 0, stream>>>(psum, pcnt, income, armies, W1f, b1f, W2f, b2f, out);

  int M = (out_size - BATCH_) / BATCH_;
  if (M < 1) M = 1;
  fill_f32_k<<<(BATCH_ * M + 255) / 256, blk, 0, stream>>>(out + BATCH_, BATCH_ * M, -logf((float)M));
}
